// HM_PGS_45217415693047
// MI455X (gfx1250) — matrix operand order repaired and checked by reading, not compiled, not run on hardware
//
#include <hip/hip_runtime.h>
#include <hip/hip_bf16.h>
#include <math.h>

// ---------------------------------------------------------------------------
// MI455X (gfx1250) implementation of the hyperbolic-GNN + GRU + transformer
// reference. All GEMM-shaped ops use v_wmma_f32_16x16x32_f16 (f32 accumulate).
// Sequential GRU recurrences run as single-workgroup persistent kernels with
// the hidden state in LDS. All buffers are L2-resident (~25MB total).
// ---------------------------------------------------------------------------

#define SS_N 974
#define HH_N 379
#define SH_N 1353
#define D_N  16
#define T_N  64
#define SQC  1.41421356237309515f

typedef __attribute__((ext_vector_type(16))) _Float16 v16h;
typedef __attribute__((ext_vector_type(8)))  float    v8f;

union U16h { v16h v; _Float16 e[16]; };
union U8f  { v8f  v; float    e[8];  };

// ---- fragment load helpers -------------------------------------------------
// 8 consecutive floats -> halves (A chunk). AV: 16B-aligned -> float4/b128.
template<bool AV>
__device__ __forceinline__ void hmpgs_frag8(const float* __restrict__ p,
                                            _Float16* dst)
{
    if (AV) {
        const float4 x0 = ((const float4*)p)[0];
        const float4 x1 = ((const float4*)p)[1];
        dst[0] = (_Float16)x0.x; dst[1] = (_Float16)x0.y;
        dst[2] = (_Float16)x0.z; dst[3] = (_Float16)x0.w;
        dst[4] = (_Float16)x1.x; dst[5] = (_Float16)x1.y;
        dst[6] = (_Float16)x1.z; dst[7] = (_Float16)x1.w;
    } else {
#pragma unroll
        for (int i = 0; i < 8; ++i) dst[i] = (_Float16)p[i];
    }
}

// 16 consecutive floats -> halves (B run, transposed-weight case).
template<bool BV>
__device__ __forceinline__ void hmpgs_frag16(const float* __restrict__ p,
                                             _Float16* dst)
{
    if (BV) {
#pragma unroll
        for (int q = 0; q < 4; ++q) {
            const float4 x = ((const float4*)p)[q];
            dst[q * 4 + 0] = (_Float16)x.x;
            dst[q * 4 + 1] = (_Float16)x.y;
            dst[q * 4 + 2] = (_Float16)x.z;
            dst[q * 4 + 3] = (_Float16)x.w;
        }
    } else {
#pragma unroll
        for (int h = 0; h < 16; ++h) dst[h] = (_Float16)p[h];
    }
}

// Guarded A fragment (edge tiles / K tail).
__device__ __forceinline__ void hmpgs_fragA_guard(const float* __restrict__ A,
                                                  int lda, int am, int M, int K,
                                                  int kbase, int hi, U16h& a)
{
#pragma unroll
    for (int h = 0; h < 16; ++h) {
        int ka = kbase + hi * 16 + h;
        a.e[h] = (_Float16)((am < M && ka < K) ? A[(size_t)am * lda + ka] : 0.f);
    }
}

// Guarded B fragment (edge tiles / K tail).
template<int BT>
__device__ __forceinline__ void hmpgs_fragB_guard(const float* __restrict__ B,
                                                  int ldb, int bn, int N, int K,
                                                  int kb0, U16h& b)
{
#pragma unroll
    for (int h = 0; h < 16; ++h) {
        int kb = kb0 + h;
        float v = 0.f;
        if (kb < K && bn < N)
            v = BT ? B[(size_t)bn * ldb + kb] : B[(size_t)kb * ldb + bn];
        b.e[h] = (_Float16)v;
    }
}

// ---------------------------------------------------------------------------
// Generic WMMA GEMM: C[M,N] = A[M,K] * B + bias (optional), f16 inputs
// converted on load, f32 accumulate.
//   BT==0: B[k][n] = Bsrc[k*ldb + n]   (B is K x N row-major)
//   BT==1: B[k][n] = Bsrc[n*ldb + k]   (Bsrc is N x K row-major, i.e. W -> W^T)
// rev_base >= 0: output row m is stored at row (rev_base - m); rows mapping to
// negative indices are dropped (implements the [::-1] time reversal).
// One wave (32 threads) per 16x16 output tile; grid = (N/16, ceil(M/16)).
// A/B fragments use contiguous k order: element s of lane half h holds k = 16h + s (wave32).
//
// The k-loop is split into an interior loop (block fully inside MxN, full
// 32-wide k-tiles: zero guards, pure b128 + cvt_pk + wmma body), an edge loop
// (per-element guarded, only the partial m/n tile rows), and one guarded
// K-tail tile. All uniform decisions are hoisted out of the loops.
// ---------------------------------------------------------------------------
template<int BT, bool AV, bool BV>
__global__ __launch_bounds__(32)
void hmpgs_wmma_gemm(const float* __restrict__ A, int lda,
                     const float* __restrict__ Bsrc, int ldb,
                     const float* __restrict__ bias,
                     float* __restrict__ C, int ldc,
                     int M, int N, int K, int rev_base)
{
    const int lane  = threadIdx.x;        // 0..31
    const int nbase = blockIdx.x * 16;
    const int mbase = blockIdx.y * 16;
    const int l15   = lane & 15;
    const int hi    = lane >> 4;          // lane half: 0 or 1

    U8f c;
#pragma unroll
    for (int r = 0; r < 8; ++r) c.e[r] = 0.f;

    const int am  = mbase + l15;          // A row for this lane
    const int bn  = nbase + l15;          // B column for this lane
    const int nKf = K >> 5;               // number of full 32-wide k-tiles
    const bool interior = (mbase + 16 <= M) && (nbase + 16 <= N);

    if (interior) {
        // ---- hot loop: no guards at all ----
        const float* pa = A + (size_t)am * lda + hi * 16;  // A chunk0 base
        for (int kt = 0; kt < nKf; ++kt) {
            const int kbase = kt << 5;
            U16h a, b;
            hmpgs_frag8<AV>(pa + kbase,      &a.e[0]);   // K = kbase + 16*hi + i
            hmpgs_frag8<AV>(pa + kbase + 8,  &a.e[8]);   // K = kbase + 16*hi + 8 + i
            if (BT) {
                hmpgs_frag16<BV>(Bsrc + (size_t)bn * ldb + kbase + hi * 16, &b.e[0]);
            } else {
                const float* pb = Bsrc + (size_t)(kbase + hi * 16) * ldb + bn;
#pragma unroll
                for (int h = 0; h < 16; ++h) b.e[h] = (_Float16)pb[(size_t)h * ldb];
            }
            c.v = __builtin_amdgcn_wmma_f32_16x16x32_f16(
                false, a.v, false, b.v, (short)0, c.v, false, false);
        }
    } else {
        // ---- edge tiles: per-element guarded ----
        for (int kt = 0; kt < nKf; ++kt) {
            const int kbase = kt << 5;
            U16h a, b;
            hmpgs_fragA_guard(A, lda, am, M, K, kbase, hi, a);
            hmpgs_fragB_guard<BT>(Bsrc, ldb, bn, N, K, kbase + hi * 16, b);
            c.v = __builtin_amdgcn_wmma_f32_16x16x32_f16(
                false, a.v, false, b.v, (short)0, c.v, false, false);
        }
    }

    if (K & 31) {
        // ---- single guarded K-tail tile (also the whole loop for K<=16) ----
        const int kbase = nKf << 5;
        U16h a, b;
        hmpgs_fragA_guard(A, lda, am, M, K, kbase, hi, a);
        hmpgs_fragB_guard<BT>(Bsrc, ldb, bn, N, K, kbase + hi * 16, b);
        c.v = __builtin_amdgcn_wmma_f32_16x16x32_f16(
            false, a.v, false, b.v, (short)0, c.v, false, false);
    }

    // C/D layout: lane%16 = N, vgpr r -> M = r + 8*lane_half
    const float bias_v = (bias && bn < N) ? bias[bn] : 0.f;
#pragma unroll
    for (int r = 0; r < 8; ++r) {
        int m = mbase + r + hi * 8;
        int outm = (rev_base >= 0) ? (rev_base - m) : m;
        if (m < M && outm >= 0 && bn < N)
            C[(size_t)outm * ldc + bn] = c.e[r] + bias_v;
    }
}

// ---------------------------------------------------------------------------
// Small helper kernels
// ---------------------------------------------------------------------------
__global__ void hmpgs_zero(float* __restrict__ p, int n)
{
    int i = blockIdx.x * blockDim.x + threadIdx.x;
    if (i < n) p[i] = 0.f;
}

// t = logmap0(expmap0(emb[idx[i]]))  (row of 16)
__global__ void hmpgs_gnn_pre(const float* __restrict__ emb,
                              const int* __restrict__ idx,
                              float* __restrict__ t, int n)
{
    int i = blockIdx.x * blockDim.x + threadIdx.x;
    if (i >= n) return;
    int row = idx ? idx[i] : i;
    float u[16], e[16];
    float nrm = 0.f;
#pragma unroll
    for (int d = 0; d < 16; ++d) { u[d] = emb[(size_t)row * 16 + d]; nrm += u[d] * u[d]; }
    nrm = fmaxf(sqrtf(nrm), 1e-7f);
    float s = tanhf(SQC * nrm) / (SQC * nrm);
    float n2 = 0.f;
#pragma unroll
    for (int d = 0; d < 16; ++d) { e[d] = s * u[d]; n2 += e[d] * e[d]; }
    n2 = fmaxf(sqrtf(n2), 1e-7f);
    float a  = fminf(SQC * n2, 1.f - 1e-5f);
    float s2 = atanhf(a) / (SQC * n2);
#pragma unroll
    for (int d = 0; d < 16; ++d) t[(size_t)i * 16 + d] = s2 * e[d];
}

// agg[dst] += t[src] over all edges (segment_sum)
__global__ void hmpgs_seg_sum(const float* __restrict__ t,
                              const int* __restrict__ edges, int E,
                              float* __restrict__ agg)
{
    int idx = blockIdx.x * blockDim.x + threadIdx.x;
    if (idx >= E * 16) return;
    int e = idx >> 4, d = idx & 15;
    int src = edges[e];
    int dst = edges[E + e];
    atomicAdd(&agg[(size_t)dst * 16 + d], t[(size_t)src * 16 + d]);
}

// h = expmap0(relu(agg @ We + be));  dec = logmap0(h) @ Wd + bd
__global__ void hmpgs_gnn_encdec(const float* __restrict__ agg,
                                 const float* __restrict__ We, const float* __restrict__ be,
                                 const float* __restrict__ Wd, const float* __restrict__ bd,
                                 float* __restrict__ h, float* __restrict__ dec, int n)
{
    int i = blockIdx.x * blockDim.x + threadIdx.x;
    if (i >= n) return;
    float a[16], y[16], hr[16], lm[16];
#pragma unroll
    for (int d = 0; d < 16; ++d) a[d] = agg[(size_t)i * 16 + d];
    float nrm = 0.f;
#pragma unroll
    for (int j = 0; j < 16; ++j) {
        float s = be[j];
#pragma unroll
        for (int k = 0; k < 16; ++k) s += a[k] * We[k * 16 + j];  // agg @ We
        y[j] = fmaxf(s, 0.f);
        nrm += y[j] * y[j];
    }
    nrm = fmaxf(sqrtf(nrm), 1e-7f);
    float se = tanhf(SQC * nrm) / (SQC * nrm);
    float n2 = 0.f;
#pragma unroll
    for (int d = 0; d < 16; ++d) { hr[d] = se * y[d]; n2 += hr[d] * hr[d]; h[(size_t)i * 16 + d] = hr[d]; }
    n2 = fmaxf(sqrtf(n2), 1e-7f);
    float sl = atanhf(fminf(SQC * n2, 1.f - 1e-5f)) / (SQC * n2);
#pragma unroll
    for (int d = 0; d < 16; ++d) lm[d] = sl * hr[d];
#pragma unroll
    for (int j = 0; j < 16; ++j) {
        float s = bd[j];
#pragma unroll
        for (int k = 0; k < 16; ++k) s += lm[k] * Wd[k * 16 + j];  // logmap0(h) @ Wd
        dec[(size_t)i * 16 + j] = s;
    }
}

// out = ((dec_sh + emb@ems_w^T + ems_b) * 0.5) @ mlp_w^T + mlp_b
__global__ void hmpgs_sh_mix(const float* __restrict__ emb, const int* __restrict__ idx,
                             const float* __restrict__ dec_sh,
                             const float* __restrict__ ems_w, const float* __restrict__ ems_b,
                             const float* __restrict__ mlp_w, const float* __restrict__ mlp_b,
                             float* __restrict__ out, int n)
{
    int i = blockIdx.x * blockDim.x + threadIdx.x;
    if (i >= n) return;
    int row = idx ? idx[i] : i;
    float x1[16], v[16];
#pragma unroll
    for (int d = 0; d < 16; ++d) x1[d] = emb[(size_t)row * 16 + d];
#pragma unroll
    for (int j = 0; j < 16; ++j) {
        float s = ems_b[j];
#pragma unroll
        for (int k = 0; k < 16; ++k) s += x1[k] * ems_w[j * 16 + k];  // x @ W^T
        v[j] = (dec_sh[(size_t)i * 16 + j] + s) * 0.5f;
    }
#pragma unroll
    for (int j = 0; j < 16; ++j) {
        float s = mlp_b[j];
#pragma unroll
        for (int k = 0; k < 16; ++k) s += v[k] * mlp_w[j * 16 + k];
        out[(size_t)i * 16 + j] = s;
    }
}

// per-column mean/var over n rows (population var), 16 columns
__global__ __launch_bounds__(256)
void hmpgs_bn_stats(const float* __restrict__ x, int n, float* __restrict__ stats)
{
    __shared__ float ssum[256], ssq[256];
    int tid = threadIdx.x;
    int col = tid & 15, grp = tid >> 4;  // 16 groups of 16 cols
    float s = 0.f, q = 0.f;
    for (int r = grp; r < n; r += 16) {
        float v = x[(size_t)r * 16 + col];
        s += v; q += v * v;
    }
    ssum[tid] = s; ssq[tid] = q;
    __syncthreads();
    if (grp == 0) {
        float S = 0.f, Q = 0.f;
        for (int g = 0; g < 16; ++g) { S += ssum[g * 16 + col]; Q += ssq[g * 16 + col]; }
        float mu = S / (float)n;
        stats[col]      = mu;
        stats[16 + col] = Q / (float)n - mu * mu;
    }
}

// es/eh 16-dim combination (batch-norm+tanh applied inline)
__global__ void hmpgs_combine(const float* __restrict__ xsh2, const float* __restrict__ stats,
                              const float* __restrict__ bng, const float* __restrict__ bnb,
                              const float* __restrict__ dec_ss, const float* __restrict__ dec_hh,
                              const float* __restrict__ Ls_w, const float* __restrict__ Ls_b,
                              const float* __restrict__ dec_ddi,
                              float* __restrict__ es, float* __restrict__ eh)
{
    int i = blockIdx.x * blockDim.x + threadIdx.x;
    if (i >= SH_N) return;
    if (i < SS_N) {
#pragma unroll
        for (int j = 0; j < 16; ++j) {
            float bn = tanhf((xsh2[(size_t)i * 16 + j] - stats[j]) * rsqrtf(stats[16 + j] + 1e-5f) * bng[j] + bnb[j]);
            es[(size_t)i * 16 + j] = (bn + dec_ss[(size_t)i * 16 + j]) * 0.5f;
        }
    } else {
        int ih = i - SS_N;
        float d[16];
#pragma unroll
        for (int k = 0; k < 16; ++k) d[k] = dec_hh[(size_t)ih * 16 + k];
#pragma unroll
        for (int j = 0; j < 16; ++j) {
            float hh2 = Ls_b[j];
#pragma unroll
            for (int k = 0; k < 16; ++k) hh2 += d[k] * Ls_w[j * 16 + k];  // dec_hh @ Ls^T
            float bn = tanhf((xsh2[(size_t)i * 16 + j] - stats[j]) * rsqrtf(stats[16 + j] + 1e-5f) * bng[j] + bnb[j]);
            eh[(size_t)ih * 16 + j] = (bn + hh2) * 0.5f - 0.1f * dec_ddi[(size_t)ih * 16 + j];
        }
    }
}

// ---------------------------------------------------------------------------
// GRU recurrence: single workgroup (512 threads = 16 waves on one WGP),
// hidden state in LDS. GI already contains Wih@x_t + bih per step.
// Next step's GI row is prefetched (global_prefetch_b8) behind the matvec.
// ---------------------------------------------------------------------------
__global__ __launch_bounds__(512)
void hmpgs_gru(const float* __restrict__ GI, const float* __restrict__ Whh,
               const float* __restrict__ bhh, float* __restrict__ outs,
               float* __restrict__ hT, int T)
{
    __shared__ float hs[512];
    int j = threadIdx.x;
    hs[j] = 0.f;
    __syncthreads();
    const float* wr = Whh + (size_t)j * 512;
    const float* wz = Whh + (size_t)(512 + j) * 512;
    const float* wn = Whh + (size_t)(1024 + j) * 512;
    const float br = bhh[j], bz = bhh[512 + j], bn = bhh[1024 + j];
    for (int t = 0; t < T; ++t) {
        if (t + 1 < T) {
            __builtin_prefetch(GI + (size_t)(t + 1) * 1536 + j, 0, 0);
            __builtin_prefetch(GI + (size_t)(t + 1) * 1536 + 512 + j, 0, 0);
            __builtin_prefetch(GI + (size_t)(t + 1) * 1536 + 1024 + j, 0, 0);
        }
        float gr = br, gz = bz, gn = bn;
        for (int k = 0; k < 512; ++k) {
            float hk = hs[k];
            gr += wr[k] * hk; gz += wz[k] * hk; gn += wn[k] * hk;
        }
        const float* gi = GI + (size_t)t * 1536;
        float r   = 1.f / (1.f + expf(-(gi[j] + gr)));
        float z   = 1.f / (1.f + expf(-(gi[512 + j] + gz)));
        float nn_ = tanhf(gi[1024 + j] + r * gn);
        float hn  = (1.f - z) * nn_ + z * hs[j];
        __syncthreads();
        hs[j] = hn;
        if (outs) outs[(size_t)t * 512 + j] = hn;
        __syncthreads();
    }
    hT[j] = hs[j];
}

// att softmax over time, o1 = att*tanh(o1pre)*i1, query = last row projection
__global__ __launch_bounds__(256)
void hmpgs_att_o1_query(const float* __restrict__ o1r,
                        const float* __restrict__ aw, const float* __restrict__ ab,
                        const float* __restrict__ o1pre, const float* __restrict__ i1,
                        const float* __restrict__ qw, const float* __restrict__ qb,
                        float* __restrict__ o1, float* __restrict__ query)
{
    __shared__ float lg[64], att[64], last[256];
    int t = threadIdx.x;
    if (t < 64) {
        float s = ab[0];
        for (int k = 0; k < 512; ++k) s += o1r[(size_t)t * 512 + k] * aw[k];
        lg[t] = s;
    }
    __syncthreads();
    if (t < 64) {
        float mx = -1e30f;
        for (int u = 0; u < 64; ++u) mx = fmaxf(mx, lg[u]);
        float sm = 0.f;
        for (int u = 0; u < 64; ++u) sm += expf(lg[u] - mx);
        att[t] = expf(lg[t] - mx) / sm;
    }
    __syncthreads();
    for (int idx = t; idx < 64 * 256; idx += 256) {
        int tt = idx >> 8, jj = idx & 255;
        float v = att[tt] * tanhf(o1pre[idx]) * i1[idx];
        o1[idx] = v;
        if (tt == 63) last[jj] = v;
    }
    __syncthreads();
    // queries = relu(concat(o1,o1)) @ qw^T + qb, last time step only.
    // concat duplicates features -> fold weights k and k+256.
    float s = qb[t];
    for (int k = 0; k < 256; ++k)
        s += fmaxf(last[k], 0.f) * (qw[(size_t)t * 512 + k] + qw[(size_t)t * 512 + 256 + k]);
    query[t] = s;
}

// emb -> self-attn (rank-1) -> LN -> FFN -> LN, all on a 64-vector
__global__ __launch_bounds__(256)
void hmpgs_enc(const float* __restrict__ h2,
               const float* __restrict__ med_w, const float* __restrict__ med_b,
               const float* __restrict__ Wv, const float* __restrict__ bv,
               const float* __restrict__ Wo, const float* __restrict__ bo,
               const float* __restrict__ ln1g, const float* __restrict__ ln1b,
               const float* __restrict__ ff1w, const float* __restrict__ ff1b,
               const float* __restrict__ ff2w, const float* __restrict__ ff2b,
               const float* __restrict__ ln2g, const float* __restrict__ ln2b,
               float* __restrict__ enc)
{
    __shared__ float e0[64], vv[64], s1[64], n1[64], ffh[2048], s2[64];
    int t = threadIdx.x;
    if (t < 64) {
        float s = med_b[t];
        for (int k = 0; k < 512; ++k) s += h2[k] * med_w[(size_t)t * 512 + k];
        e0[t] = s;
    }
    __syncthreads();
    if (t < 64) {
        float s = bv[t];
        for (int k = 0; k < 64; ++k) s += e0[k] * Wv[t * 64 + k];
        vv[t] = s;
    }
    __syncthreads();
    if (t < 64) {
        float s = bo[t];
        for (int k = 0; k < 64; ++k) s += vv[k] * Wo[t * 64 + k];
        s1[t] = e0[t] + s;           // emb + sa
    }
    __syncthreads();
    if (t < 64) {
        float m = 0.f;
        for (int u = 0; u < 64; ++u) m += s1[u];
        m *= (1.f / 64.f);
        float v = 0.f;
        for (int u = 0; u < 64; ++u) { float d = s1[u] - m; v += d * d; }
        v *= (1.f / 64.f);
        n1[t] = (s1[t] - m) * rsqrtf(v + 1e-5f) * ln1g[t] + ln1b[t];
    }
    __syncthreads();
    for (int n = t; n < 2048; n += 256) {
        float s = ff1b[n];
        for (int k = 0; k < 64; ++k) s += n1[k] * ff1w[(size_t)n * 64 + k];
        ffh[n] = fmaxf(s, 0.f);
    }
    __syncthreads();
    if (t < 64) {
        float s = ff2b[t];
        for (int k = 0; k < 2048; ++k) s += ffh[k] * ff2w[(size_t)t * 2048 + k];
        s2[t] = n1[t] + s;
    }
    __syncthreads();
    if (t < 64) {
        float m = 0.f;
        for (int u = 0; u < 64; ++u) m += s2[u];
        m *= (1.f / 64.f);
        float v = 0.f;
        for (int u = 0; u < 64; ++u) { float d = s2[u] - m; v += d * d; }
        v *= (1.f / 64.f);
        enc[t] = (s2[t] - m) * rsqrtf(v + 1e-5f) * ln2g[t] + ln2b[t];
    }
}

// b = tile(enc, 64).flatten @ tran_w^T + tran_b
// (prob_g is exactly all-ones: keys are zero -> softmax over a size-1 axis
//  gives 1.0, and softmax(1.0) over size-1 axis is again 1.0)
__global__ __launch_bounds__(256)
void hmpgs_bvec(const float* __restrict__ enc, const float* __restrict__ tranw,
                const float* __restrict__ tranb, float* __restrict__ bvec)
{
    __shared__ float e[64];
    int t = threadIdx.x;
    if (t < 64) e[t] = enc[t];
    __syncthreads();
    int n = blockIdx.x * 256 + t;
    if (n < HH_N) {
        float s = tranb[n];
        const float* row = tranw + (size_t)n * 4096;
        for (int m = 0; m < 4096; ++m) s += e[m & 63] * row[m];
        bvec[n] = s;
    }
}

// fun2 / softmax-attn / out MLP / final mix + fin_e + fin_d
__global__ __launch_bounds__(512)
void hmpgs_final(const float* __restrict__ EH, const float* __restrict__ query,
                 const float* __restrict__ bvec,
                 const float* __restrict__ out1w, const float* __restrict__ out1b,
                 const float* __restrict__ out2w, const float* __restrict__ out2b,
                 const float* __restrict__ samplew, const float* __restrict__ sampleb,
                 const float* __restrict__ sample2w, const float* __restrict__ sample2b,
                 const float* __restrict__ hddi, float* __restrict__ out)
{
    __shared__ float q[256], fun[HH_N], w1[HH_N], sm[HH_N], fact[256], h1[32];
    int t = threadIdx.x;
    if (t < 256) q[t] = query[t];
    __syncthreads();
    if (t < HH_N) {
        float s = 0.f;
        for (int k = 0; k < 256; ++k) s += q[k] * EH[(size_t)t * 256 + k];
        fun[t] = s;
        w1[t] = 1.f / (1.f + expf(-s));
    }
    __syncthreads();
    if (t < HH_N) {
        float mx = -1e30f;
        for (int u = 0; u < HH_N; ++u) mx = fmaxf(mx, fun[u]);
        float ss = 0.f;
        for (int u = 0; u < HH_N; ++u) ss += expf(fun[u] - mx);
        sm[t] = expf(fun[t] - mx) / ss;
    }
    __syncthreads();
    if (t < 256) {
        float s = 0.f;
        for (int j = 0; j < HH_N; ++j) s += sm[j] * EH[(size_t)j * 256 + t];
        fact[t] = s;
    }
    __syncthreads();
    if (t < 32) {
        float s = out1b[t];
        for (int k = 0; k < 256; ++k) s += fmaxf(q[k], 0.f)    * out1w[(size_t)t * 512 + k];
        for (int k = 0; k < 256; ++k) s += fmaxf(fact[k], 0.f) * out1w[(size_t)t * 512 + 256 + k];
        h1[t] = fmaxf(s, 0.f);
    }
    __syncthreads();
    if (t < HH_N) {
        float s = out2b[t];
        for (int m = 0; m < 32; ++m) s += h1[m] * out2w[t * 32 + m];
        out[t] = w1[t] * (0.9f * s) + 1.85f * bvec[t];
        float fe = sampleb[0];
        for (int k = 0; k < 256; ++k) fe += EH[(size_t)t * 256 + k] * samplew[k];
        out[HH_N + t] = fe;
        float fd = sample2b[0];
        for (int d = 0; d < 16; ++d) fd += hddi[(size_t)t * 16 + d] * sample2w[d];
        out[2 * HH_N + t] = fd;
    }
}

// ---------------------------------------------------------------------------
// Host launcher. Input index map assumes setup_inputs() dict-insertion-order
// flattening (params dict flattened in insertion order after the 9 arrays).
// ---------------------------------------------------------------------------
extern "C" void kernel_launch(void* const* d_in, const int* in_sizes, int n_in,
                              void* d_out, int out_size, void* d_ws, size_t ws_size,
                              hipStream_t stream)
{
    (void)n_in; (void)out_size; (void)ws_size;
    const int* xSH  = (const int*)d_in[0];
    const int* xSS  = (const int*)d_in[1];
    const int* xHH  = (const int*)d_in[2];
    const int* eSH  = (const int*)d_in[3];
    const int* eSS  = (const int*)d_in[4];
    const int* eHH  = (const int*)d_in[5];
    const int* eDDI = (const int*)d_in[6];
    const float* diag = (const float*)d_in[7];   // (64, 974)
    const float* med  = (const float*)d_in[8];   // (64, 379)
    const float* P9   = (const float*)d_in[9];   // emb_SH (1353,16)
    const float* P10  = (const float*)d_in[10];  // emb_SS
    const float* P11  = (const float*)d_in[11];  // emb_HH
    auto F = [&](int i) { return (const float*)d_in[i]; };

    const int EshN  = in_sizes[3] / 2;
    const int EssN  = in_sizes[4] / 2;
    const int EhhN  = in_sizes[5] / 2;
    const int EddiN = in_sizes[6] / 2;

    // ---- workspace layout (floats; every size a multiple of 16 -> 64B align) ----
    float* ws = (float*)d_ws;
    size_t off = 0;
    auto WSA = [&](size_t n) { float* p = ws + off; off += n; return p; };
    float* tSH   = WSA((size_t)SH_N * 16);
    float* tSS   = WSA((size_t)SS_N * 16);
    float* tHH   = WSA((size_t)HH_N * 16);
    float* aggSH = WSA((size_t)SH_N * 16);   // aggs contiguous: zero once
    float* aggSS = WSA((size_t)SS_N * 16);
    float* aggHH = WSA((size_t)HH_N * 16);
    float* aggDD = WSA((size_t)HH_N * 16);
    float* hSH   = WSA((size_t)SH_N * 16);  float* decSH = WSA((size_t)SH_N * 16);
    float* hSS   = WSA((size_t)SS_N * 16);  float* decSS = WSA((size_t)SS_N * 16);
    float* hHH   = WSA((size_t)HH_N * 16);  float* decHH = WSA((size_t)HH_N * 16);
    float* hDDI  = WSA((size_t)HH_N * 16);  float* decDD = WSA((size_t)HH_N * 16);
    float* xsh2  = WSA((size_t)SH_N * 16);
    float* stats = WSA(32);
    float* es16  = WSA((size_t)SS_N * 16);
    float* eh16  = WSA((size_t)HH_N * 16);
    float* ES    = WSA((size_t)SS_N * 256);
    float* EH    = WSA((size_t)HH_N * 256);
    float* I1    = WSA((size_t)64 * 256);    // already time-reversed
    float* I2    = WSA((size_t)64 * 256);    // rows 0..62 valid, reversed
    float* GI0   = WSA((size_t)64 * 1536);
    float* GI1   = WSA((size_t)64 * 1536);
    float* O1R   = WSA((size_t)64 * 512);
    float* O1P   = WSA((size_t)64 * 256);
    float* O1    = WSA((size_t)64 * 256);
    float* QRY   = WSA(256);
    float* H2    = WSA(512);
    float* HTD   = WSA(512);
    float* ENC   = WSA(64);
    float* BV    = WSA(HH_N + 5);

    const int B = 256;
    auto G = [&](int n) { return dim3((n + B - 1) / B); };

    // ---- GNN stage ----
    int aggTot = (SH_N + SS_N + HH_N + HH_N) * 16;
    hmpgs_zero<<<G(aggTot), B, 0, stream>>>(aggSH, aggTot);
    hmpgs_gnn_pre<<<G(SH_N), B, 0, stream>>>(P9,  xSH, tSH, SH_N);
    hmpgs_gnn_pre<<<G(SS_N), B, 0, stream>>>(P10, xSS, tSS, SS_N);
    hmpgs_gnn_pre<<<G(HH_N), B, 0, stream>>>(P11, xHH, tHH, HH_N);
    hmpgs_seg_sum<<<G(EshN  * 16), B, 0, stream>>>(tSH, eSH,  EshN,  aggSH);
    hmpgs_seg_sum<<<G(EssN  * 16), B, 0, stream>>>(tSS, eSS,  EssN,  aggSS);
    hmpgs_seg_sum<<<G(EhhN  * 16), B, 0, stream>>>(tHH, eHH,  EhhN,  aggHH);
    hmpgs_seg_sum<<<G(EddiN * 16), B, 0, stream>>>(tHH, eDDI, EddiN, aggDD);
    hmpgs_gnn_encdec<<<G(SH_N), B, 0, stream>>>(aggSH, F(12), F(13), F(14), F(15), hSH,  decSH, SH_N);
    hmpgs_gnn_encdec<<<G(SS_N), B, 0, stream>>>(aggSS, F(16), F(17), F(18), F(19), hSS,  decSS, SS_N);
    hmpgs_gnn_encdec<<<G(HH_N), B, 0, stream>>>(aggHH, F(20), F(21), F(22), F(23), hHH,  decHH, HH_N);
    hmpgs_gnn_encdec<<<G(HH_N), B, 0, stream>>>(aggDD, F(24), F(25), F(26), F(27), hDDI, decDD, HH_N);

    // ---- SH mix, batch-norm, es/eh combine ----
    hmpgs_sh_mix<<<G(SH_N), B, 0, stream>>>(P9, xSH, decSH, F(28), F(29), F(30), F(31), xsh2, SH_N);
    hmpgs_bn_stats<<<1, 256, 0, stream>>>(xsh2, SH_N, stats);
    hmpgs_combine<<<G(SH_N), B, 0, stream>>>(xsh2, stats, F(52), F(53), decSS, decHH,
                                             F(32), F(33), decDD, es16, eh16);

    // ---- WMMA GEMMs ----
    // ES = es16 @ xs_w^T + xs_b          (974,16)x(16,256)
    hmpgs_wmma_gemm<1, true, true><<<dim3(16, 61), 32, 0, stream>>>(
        es16, 16, F(34), 16, F(35), ES, 256, SS_N, 256, 16, -1);
    // EH = eh16 @ xh_w^T + xh_b          (379,16)x(16,256)
    hmpgs_wmma_gemm<1, true, true><<<dim3(16, 24), 32, 0, stream>>>(
        eh16, 16, F(36), 16, F(37), EH, 256, HH_N, 256, 16, -1);
    // I1 = reverse(diag @ ES)            (64,974)x(974,256), rows stored reversed
    hmpgs_wmma_gemm<0, false, true><<<dim3(16, 4), 32, 0, stream>>>(
        diag, SS_N, ES, 256, nullptr, I1, 256, 64, 256, SS_N, 63);
    // I2 = reverse((med @ EH)[:63])      rows 0..62 reversed, row 63 dropped
    hmpgs_wmma_gemm<0, false, true><<<dim3(16, 4), 32, 0, stream>>>(
        med, HH_N, EH, 256, nullptr, I2, 256, 64, 256, HH_N, 62);
    // GI0 = I1 @ r0_Wih^T + r0_bih       (64,256)x(256,1536)
    hmpgs_wmma_gemm<1, true, true><<<dim3(96, 4), 32, 0, stream>>>(
        I1, 256, F(54), 256, F(56), GI0, 1536, 64, 1536, 256, -1);
    // GI1 = I2 @ r1_Wih^T + r1_bih       (63,256)x(256,1536)
    hmpgs_wmma_gemm<1, true, true><<<dim3(96, 4), 32, 0, stream>>>(
        I2, 256, F(58), 256, F(60), GI1, 1536, 63, 1536, 256, -1);

    // ---- GRU r0 + attention gate ----
    hmpgs_gru<<<1, 512, 0, stream>>>(GI0, F(55), F(57), O1R, HTD, 64);
    // O1P = o1r @ r0_bw^T + r0_bb        (64,512)x(512,256)
    hmpgs_wmma_gemm<1, true, true><<<dim3(16, 4), 32, 0, stream>>>(
        O1R, 512, F(64), 512, F(65), O1P, 256, 64, 256, 512, -1);
    hmpgs_att_o1_query<<<1, 256, 0, stream>>>(O1R, F(62), F(63), O1P, I1,
                                              F(38), F(39), O1, QRY);

    // ---- GRU r1 + encoder block ----
    hmpgs_gru<<<1, 512, 0, stream>>>(GI1, F(59), F(61), nullptr, H2, 63);
    hmpgs_enc<<<1, 256, 0, stream>>>(H2, F(40), F(41), F(66), F(67), F(68), F(69),
                                     F(70), F(71), F(72), F(73), F(74), F(75),
                                     F(76), F(77), ENC);
    hmpgs_bvec<<<2, 256, 0, stream>>>(ENC, F(42), F(43), BV);

    // ---- final fused head: a, fin_e, fin_d ----
    hmpgs_final<<<1, 512, 0, stream>>>(EH, QRY, BV, F(44), F(45), F(46), F(47),
                                       F(48), F(49), F(50), F(51), hDDI,
                                       (float*)d_out);
}
